// GRU_58832462021369
// MI455X (gfx1250) — matrix operand order repaired and checked by reading, not compiled, not run on hardware
//
#include <hip/hip_runtime.h>

#define SEQ   64
#define BATCH 64
#define HID   1024
#define EMB   512
#define VOC   10000
#define ROWS  (SEQ * BATCH) /* 4096 */

typedef __bf16 bf16_t;
typedef bf16_t v16bf __attribute__((ext_vector_type(16)));
typedef bf16_t v8bf  __attribute__((ext_vector_type(8)));
typedef float  v8f   __attribute__((ext_vector_type(8)));

// ---------------------------------------------------------------------------
// WMMA helpers (CDNA5 wave32, 16x16x32 bf16 -> f32)
// ---------------------------------------------------------------------------
__device__ __forceinline__ v8f wmma_bf16(v16bf a, v16bf b, v8f c) {
  return __builtin_amdgcn_wmma_f32_16x16x32_bf16(
      /*neg_a=*/false, a, /*neg_b=*/false, b,
      /*c_mod=*/(short)0, c, /*reuse_a=*/false, /*reuse_b=*/false);
}

// A fragment (16x32 bf16) from a bf16 row-major matrix.
// k order used here (contiguous, as in load_b_tr): lane l holds row M=l&15; lanes 0-15: K=0..15,
// lanes 16-31: K=16..31 (relative to kbase) -> two 16B loads.
__device__ __forceinline__ v16bf load_a_bf16(const bf16_t* __restrict__ A, int lda,
                                             int mbase, int kbase, int lane) {
  const int m = mbase + (lane & 15);
  const int kofs = (lane < 16) ? 0 : 16;
  const bf16_t* p = A + (size_t)m * lda + kbase + kofs;
  v8bf lo = *reinterpret_cast<const v8bf*>(p);
  v8bf hi = *reinterpret_cast<const v8bf*>(p + 8);
  return __builtin_shufflevector(lo, hi, 0, 1, 2, 3, 4, 5, 6, 7,
                                 8, 9, 10, 11, 12, 13, 14, 15);
}

// B fragment (32x16 bf16) from TRANSPOSED weight Wt[n][k] (k contiguous).
// Lane l holds col N=l&15; lanes 0-15: K=kbase+0..15, lanes 16-31:
// K=kbase+16..31 -> one aligned 32-byte load per lane.
__device__ __forceinline__ v16bf load_b_tr(const bf16_t* __restrict__ Wt, int ldk,
                                           int kbase, int nbase, int lane) {
  const int n = nbase + (lane & 15);
  const int k = kbase + ((lane < 16) ? 0 : 16);
  return *reinterpret_cast<const v16bf*>(Wt + (size_t)n * ldk + k);
}

// Virtual concatenation [X (64xHID) | Hpart (64xHID)] as A source (bf16).
__device__ __forceinline__ v16bf load_a_split(const bf16_t* __restrict__ X,
                                              const bf16_t* __restrict__ Hp,
                                              int mbase, int kbase, int lane) {
  if (kbase < HID) return load_a_bf16(X, HID, mbase, kbase, lane);
  return load_a_bf16(Hp, HID, mbase, kbase - HID, lane);
}

__device__ __forceinline__ float sigmoidf_(float x) {
  return 1.0f / (1.0f + __expf(-x));
}

// ---------------------------------------------------------------------------
// Prep kernels
// ---------------------------------------------------------------------------

// W fp32 row-major [rows][N] (K rows from rowofs) -> Wt bf16 [N][K].
__global__ void cvt_w_tr(const float* __restrict__ W, bf16_t* __restrict__ Wt,
                         int N, int rowofs, int K) {
  long long i = (long long)blockIdx.x * 256 + threadIdx.x;
  if (i >= (long long)N * K) return;
  int n = (int)(i / K);
  int k = (int)(i % K);
  Wt[i] = (bf16_t)W[(size_t)(rowofs + k) * N + n];
}

// Embedding gather, converted straight to bf16 (A-side of pre-GEMM).
__global__ void gather_emb(const int* __restrict__ inputs,
                           const float* __restrict__ table,
                           bf16_t* __restrict__ Aemb) {
  int r = blockIdx.x;                      // 0..ROWS-1  (= s*B + b)
  int e = blockIdx.y * 256 + threadIdx.x;  // 0..EMB-1
  Aemb[(size_t)r * EMB + e] = (bf16_t)table[(size_t)inputs[r] * EMB + e];
}

__global__ void init_h(const float* __restrict__ init_hidden,
                       float* __restrict__ h0, float* __restrict__ h1,
                       bf16_t* __restrict__ h0b, bf16_t* __restrict__ h1b) {
  int i = blockIdx.x * 256 + threadIdx.x;  // 0..BATCH*HID-1
  float a = init_hidden[i];
  float b = init_hidden[BATCH * HID + i];
  h0[i] = a; h1[i] = b;
  h0b[i] = (bf16_t)a; h1b[i] = (bf16_t)b;
}

// Pre-GEMM: Xg[4096][HID] = emb @ Wgx^T, g in {r,z,h}. Tm=4 per wave.
__global__ void __launch_bounds__(32)
preX_gemm(const bf16_t* __restrict__ Aemb, const bf16_t* __restrict__ bWx_all,
          float* __restrict__ X_all) {
  const int lane = threadIdx.x;
  const int mb = blockIdx.x * 64;   // 64 groups of 4 m-tiles
  const int nb = blockIdx.y * 16;   // 64 n-tiles
  const int g  = blockIdx.z;
  const bf16_t* Wt = bWx_all + (size_t)g * HID * EMB;
  float* X = X_all + (size_t)g * ROWS * HID;
  v8f acc[4] = {};
  for (int k = 0; k < EMB; k += 32) {
    v16bf b = load_b_tr(Wt, EMB, k, nb, lane);
#pragma unroll
    for (int mi = 0; mi < 4; ++mi) {
      v16bf a = load_a_bf16(Aemb, EMB, mb + mi * 16, k, lane);
      acc[mi] = wmma_bf16(a, b, acc[mi]);
    }
  }
  const int n = nb + (lane & 15);
  const int mo = (lane < 16) ? 0 : 8;
#pragma unroll
  for (int mi = 0; mi < 4; ++mi)
#pragma unroll
    for (int v = 0; v < 8; ++v)
      X[(size_t)(mb + mi * 16 + mo + v) * HID + n] = acc[mi][v];
}

// ---------------------------------------------------------------------------
// Per-timestep kernels: grid = (HID/16) blocks, 1 wave each, Tm=4 covers M=64.
// Weights are read exactly once per step (max B-reuse across the batch).
// ---------------------------------------------------------------------------

// Layer 0 phase 1: r,z gates; emit rh = r*h_old (bf16) and z (f32).
__global__ void __launch_bounds__(32)
gru0_rz(const bf16_t* __restrict__ hb_old, const float* __restrict__ h_old,
        const bf16_t* __restrict__ bWrh, const bf16_t* __restrict__ bWzh,
        const float* __restrict__ Xr, const float* __restrict__ Xz,
        const float* __restrict__ br, const float* __restrict__ bz,
        bf16_t* __restrict__ rhb, float* __restrict__ zbuf, int t) {
  const int lane = threadIdx.x;
  const int nb = blockIdx.x * 16;
  v8f ar[4] = {}, az[4] = {};
  for (int k = 0; k < HID; k += 32) {
    v16bf brv = load_b_tr(bWrh, HID, k, nb, lane);
    v16bf bzv = load_b_tr(bWzh, HID, k, nb, lane);
#pragma unroll
    for (int mi = 0; mi < 4; ++mi) {
      v16bf a = load_a_bf16(hb_old, HID, mi * 16, k, lane);
      ar[mi] = wmma_bf16(a, brv, ar[mi]);
      az[mi] = wmma_bf16(a, bzv, az[mi]);
    }
  }
  const int n = nb + (lane & 15);
  const int mo = (lane < 16) ? 0 : 8;
#pragma unroll
  for (int mi = 0; mi < 4; ++mi)
#pragma unroll
    for (int v = 0; v < 8; ++v) {
      const int m = mi * 16 + mo + v;
      const size_t gi = (size_t)m * HID + n;
      const size_t xi = (size_t)(t * BATCH + m) * HID + n;
      float r = sigmoidf_(ar[mi][v] + Xr[xi] + br[n]);
      float z = sigmoidf_(az[mi][v] + Xz[xi] + bz[n]);
      rhb[gi] = (bf16_t)(r * h_old[gi]);
      zbuf[gi] = z;
    }
}

// Layer 0 phase 2: h_hat = sigmoid(Xh + rh@Whh + bh); h_new = (1-z)h + z*h_hat.
__global__ void __launch_bounds__(32)
gru0_h(const float* __restrict__ h_old, const bf16_t* __restrict__ rhb,
       const bf16_t* __restrict__ bWhh, const float* __restrict__ Xh,
       const float* __restrict__ bh, const float* __restrict__ zbuf,
       float* __restrict__ h_new, bf16_t* __restrict__ hb_new, int t) {
  const int lane = threadIdx.x;
  const int nb = blockIdx.x * 16;
  v8f ah[4] = {};
  for (int k = 0; k < HID; k += 32) {
    v16bf bv = load_b_tr(bWhh, HID, k, nb, lane);
#pragma unroll
    for (int mi = 0; mi < 4; ++mi) {
      v16bf a = load_a_bf16(rhb, HID, mi * 16, k, lane);
      ah[mi] = wmma_bf16(a, bv, ah[mi]);
    }
  }
  const int n = nb + (lane & 15);
  const int mo = (lane < 16) ? 0 : 8;
#pragma unroll
  for (int mi = 0; mi < 4; ++mi)
#pragma unroll
    for (int v = 0; v < 8; ++v) {
      const int m = mi * 16 + mo + v;
      const size_t gi = (size_t)m * HID + n;
      const size_t xi = (size_t)(t * BATCH + m) * HID + n;
      float hh = sigmoidf_(ah[mi][v] + Xh[xi] + bh[n]);
      float z = zbuf[gi];
      float hn = (1.0f - z) * h_old[gi] + z * hh;
      h_new[gi] = hn;
      hb_new[gi] = (bf16_t)hn;
    }
}

// Inter-layer FC: x1 = sigmoid(h0_new @ Wfc + bfc) -> bf16 only.
__global__ void __launch_bounds__(32)
fc_sig(const bf16_t* __restrict__ hbin, const bf16_t* __restrict__ bWfc,
       const float* __restrict__ bfc, bf16_t* __restrict__ x1b) {
  const int lane = threadIdx.x;
  const int nb = blockIdx.x * 16;
  v8f acc[4] = {};
  for (int k = 0; k < HID; k += 32) {
    v16bf bv = load_b_tr(bWfc, HID, k, nb, lane);
#pragma unroll
    for (int mi = 0; mi < 4; ++mi) {
      v16bf a = load_a_bf16(hbin, HID, mi * 16, k, lane);
      acc[mi] = wmma_bf16(a, bv, acc[mi]);
    }
  }
  const int n = nb + (lane & 15);
  const int mo = (lane < 16) ? 0 : 8;
  const float bb = bfc[n];
#pragma unroll
  for (int mi = 0; mi < 4; ++mi)
#pragma unroll
    for (int v = 0; v < 8; ++v)
      x1b[(size_t)(mi * 16 + mo + v) * HID + n] = (bf16_t)sigmoidf_(acc[mi][v] + bb);
}

// Layer 1 phase 1: r,z over K=2*HID with A = [x1 | h_old].
__global__ void __launch_bounds__(32)
gru1_rz(const bf16_t* __restrict__ x1b, const bf16_t* __restrict__ h1b_old,
        const float* __restrict__ h1_old,
        const bf16_t* __restrict__ bWr, const bf16_t* __restrict__ bWz,
        const float* __restrict__ br, const float* __restrict__ bz,
        bf16_t* __restrict__ rhb, float* __restrict__ zbuf) {
  const int lane = threadIdx.x;
  const int nb = blockIdx.x * 16;
  v8f ar[4] = {}, az[4] = {};
  for (int k = 0; k < 2 * HID; k += 32) {
    v16bf brv = load_b_tr(bWr, 2 * HID, k, nb, lane);
    v16bf bzv = load_b_tr(bWz, 2 * HID, k, nb, lane);
#pragma unroll
    for (int mi = 0; mi < 4; ++mi) {
      v16bf a = load_a_split(x1b, h1b_old, mi * 16, k, lane);
      ar[mi] = wmma_bf16(a, brv, ar[mi]);
      az[mi] = wmma_bf16(a, bzv, az[mi]);
    }
  }
  const int n = nb + (lane & 15);
  const int mo = (lane < 16) ? 0 : 8;
#pragma unroll
  for (int mi = 0; mi < 4; ++mi)
#pragma unroll
    for (int v = 0; v < 8; ++v) {
      const int m = mi * 16 + mo + v;
      const size_t gi = (size_t)m * HID + n;
      float r = sigmoidf_(ar[mi][v] + br[n]);
      float z = sigmoidf_(az[mi][v] + bz[n]);
      rhb[gi] = (bf16_t)(r * h1_old[gi]);
      zbuf[gi] = z;
    }
}

// Layer 1 phase 2: h_hat over A = [x1 | r*h]; update h; stash bf16 H1 row.
__global__ void __launch_bounds__(32)
gru1_h(const bf16_t* __restrict__ x1b, const bf16_t* __restrict__ rhb,
       const float* __restrict__ h1_old, const bf16_t* __restrict__ bWh,
       const float* __restrict__ bh, const float* __restrict__ zbuf,
       float* __restrict__ h_new, bf16_t* __restrict__ hb_new,
       bf16_t* __restrict__ H1bf, int t) {
  const int lane = threadIdx.x;
  const int nb = blockIdx.x * 16;
  v8f ah[4] = {};
  for (int k = 0; k < 2 * HID; k += 32) {
    v16bf bv = load_b_tr(bWh, 2 * HID, k, nb, lane);
#pragma unroll
    for (int mi = 0; mi < 4; ++mi) {
      v16bf a = load_a_split(x1b, rhb, mi * 16, k, lane);
      ah[mi] = wmma_bf16(a, bv, ah[mi]);
    }
  }
  const int n = nb + (lane & 15);
  const int mo = (lane < 16) ? 0 : 8;
#pragma unroll
  for (int mi = 0; mi < 4; ++mi)
#pragma unroll
    for (int v = 0; v < 8; ++v) {
      const int m = mi * 16 + mo + v;
      const size_t gi = (size_t)m * HID + n;
      float hh = sigmoidf_(ah[mi][v] + bh[n]);
      float z = zbuf[gi];
      float hn = (1.0f - z) * h1_old[gi] + z * hh;
      h_new[gi] = hn;
      hb_new[gi] = (bf16_t)hn;
      H1bf[(size_t)(t * BATCH + m) * HID + n] = (bf16_t)hn;
    }
}

// ---------------------------------------------------------------------------
// Output projection: logits[4096][VOC] = H1bf @ Wout^T + bout (84 GFLOP).
// Register-blocked Tm=2 x Tn=5 (625 n-tiles divide exactly by 5).
// ---------------------------------------------------------------------------
__global__ void __launch_bounds__(32)
out_gemm(const bf16_t* __restrict__ H1bf, const bf16_t* __restrict__ bWout,
         const float* __restrict__ bout, float* __restrict__ out) {
  const int lane = threadIdx.x;
  const int mb = blockIdx.x * 32;   // 128 groups of 2 m-tiles
  const int nb = blockIdx.y * 80;   // 125 groups of 5 n-tiles
  v8f acc[2][5] = {};
  for (int k = 0; k < HID; k += 32) {
    v16bf a0 = load_a_bf16(H1bf, HID, mb, k, lane);
    v16bf a1 = load_a_bf16(H1bf, HID, mb + 16, k, lane);
#pragma unroll
    for (int nj = 0; nj < 5; ++nj) {
      v16bf b = load_b_tr(bWout, HID, k, nb + nj * 16, lane);
      acc[0][nj] = wmma_bf16(a0, b, acc[0][nj]);
      acc[1][nj] = wmma_bf16(a1, b, acc[1][nj]);
    }
  }
  const int mo = (lane < 16) ? 0 : 8;
#pragma unroll
  for (int nj = 0; nj < 5; ++nj) {
    const int n = nb + nj * 16 + (lane & 15);
    const float bb = bout[n];
#pragma unroll
    for (int mi = 0; mi < 2; ++mi)
#pragma unroll
      for (int v = 0; v < 8; ++v)
        out[(size_t)(mb + mi * 16 + mo + v) * VOC + n] = acc[mi][nj][v] + bb;
  }
}

__global__ void copy_hfinal(const float* __restrict__ h0,
                            const float* __restrict__ h1,
                            float* __restrict__ out) {
  int i = blockIdx.x * 256 + threadIdx.x;  // 0..BATCH*HID-1
  out[i] = h0[i];
  out[BATCH * HID + i] = h1[i];
}

// ---------------------------------------------------------------------------
// Host-side orchestration
// ---------------------------------------------------------------------------
extern "C" void kernel_launch(void* const* d_in, const int* in_sizes, int n_in,
                              void* d_out, int out_size, void* d_ws, size_t ws_size,
                              hipStream_t stream) {
  (void)in_sizes; (void)n_in; (void)out_size; (void)ws_size;

  const int*   inputs      = (const int*)  d_in[0];
  const float* init_hidden = (const float*)d_in[1];
  const float* emb_table   = (const float*)d_in[2];
  const float* Wr0 = (const float*)d_in[3];
  const float* Wz0 = (const float*)d_in[4];
  const float* Wh0 = (const float*)d_in[5];
  const float* br0 = (const float*)d_in[6];
  const float* bz0 = (const float*)d_in[7];
  const float* bh0 = (const float*)d_in[8];
  const float* Wr1 = (const float*)d_in[9];
  const float* Wz1 = (const float*)d_in[10];
  const float* Wh1 = (const float*)d_in[11];
  const float* br1 = (const float*)d_in[12];
  const float* bz1 = (const float*)d_in[13];
  const float* bh1 = (const float*)d_in[14];
  const float* Wfc0 = (const float*)d_in[15];
  const float* bfc0 = (const float*)d_in[16];
  const float* Wout = (const float*)d_in[17];
  const float* bout = (const float*)d_in[18];

  // ---- workspace carve-up ----
  char* ws = (char*)d_ws;
  size_t off = 0;
  auto take = [&](size_t bytes) -> void* {
    void* p = ws + off;
    off += (bytes + 255) & ~(size_t)255;
    return p;
  };
  bf16_t* Aemb    = (bf16_t*)take((size_t)ROWS * EMB * 2);
  float*  Xall    = (float*) take((size_t)3 * ROWS * HID * 4);
  bf16_t* bWx_all = (bf16_t*)take((size_t)3 * HID * EMB * 2);
  bf16_t* bWrh    = (bf16_t*)take((size_t)HID * HID * 2);
  bf16_t* bWzh    = (bf16_t*)take((size_t)HID * HID * 2);
  bf16_t* bWhh    = (bf16_t*)take((size_t)HID * HID * 2);
  bf16_t* bWfc    = (bf16_t*)take((size_t)HID * HID * 2);
  bf16_t* bWr1    = (bf16_t*)take((size_t)HID * 2 * HID * 2);
  bf16_t* bWz1    = (bf16_t*)take((size_t)HID * 2 * HID * 2);
  bf16_t* bWh1    = (bf16_t*)take((size_t)HID * 2 * HID * 2);
  bf16_t* bWoutT  = (bf16_t*)take((size_t)VOC * HID * 2);
  bf16_t* H1bf    = (bf16_t*)take((size_t)ROWS * HID * 2);
  float*  h0p[2]  = {(float*)take((size_t)BATCH * HID * 4),
                     (float*)take((size_t)BATCH * HID * 4)};
  float*  h1p[2]  = {(float*)take((size_t)BATCH * HID * 4),
                     (float*)take((size_t)BATCH * HID * 4)};
  bf16_t* h0bp[2] = {(bf16_t*)take((size_t)BATCH * HID * 2),
                     (bf16_t*)take((size_t)BATCH * HID * 2)};
  bf16_t* h1bp[2] = {(bf16_t*)take((size_t)BATCH * HID * 2),
                     (bf16_t*)take((size_t)BATCH * HID * 2)};
  bf16_t* x1b     = (bf16_t*)take((size_t)BATCH * HID * 2);
  bf16_t* rhb     = (bf16_t*)take((size_t)BATCH * HID * 2);
  float*  zb      = (float*) take((size_t)BATCH * HID * 4);

  // ---- weight transpose + bf16 convert ----
  auto cvt = [&](const float* W, bf16_t* Wt, int N, int rowofs, int K) {
    long long tot = (long long)N * K;
    int blocks = (int)((tot + 255) / 256);
    cvt_w_tr<<<blocks, 256, 0, stream>>>(W, Wt, N, rowofs, K);
  };
  cvt(Wr0, bWx_all + (size_t)0 * HID * EMB, HID, 0, EMB);
  cvt(Wz0, bWx_all + (size_t)1 * HID * EMB, HID, 0, EMB);
  cvt(Wh0, bWx_all + (size_t)2 * HID * EMB, HID, 0, EMB);
  cvt(Wr0, bWrh, HID, EMB, HID);
  cvt(Wz0, bWzh, HID, EMB, HID);
  cvt(Wh0, bWhh, HID, EMB, HID);
  cvt(Wfc0, bWfc, HID, 0, HID);
  cvt(Wr1, bWr1, HID, 0, 2 * HID);
  cvt(Wz1, bWz1, HID, 0, 2 * HID);
  cvt(Wh1, bWh1, HID, 0, 2 * HID);
  cvt(Wout, bWoutT, VOC, 0, HID);

  // ---- embedding gather + hidden init ----
  gather_emb<<<dim3(ROWS, EMB / 256), 256, 0, stream>>>(inputs, emb_table, Aemb);
  init_h<<<(BATCH * HID) / 256, 256, 0, stream>>>(init_hidden, h0p[0], h1p[0],
                                                  h0bp[0], h1bp[0]);

  // ---- hoisted input-side gate GEMMs (all timesteps at once) ----
  preX_gemm<<<dim3(ROWS / 64, HID / 16, 3), 32, 0, stream>>>(Aemb, bWx_all, Xall);
  const float* Xr = Xall + (size_t)0 * ROWS * HID;
  const float* Xz = Xall + (size_t)1 * ROWS * HID;
  const float* Xh = Xall + (size_t)2 * ROWS * HID;

  // ---- sequential scan over time ----
  const dim3 g(HID / 16);
  for (int t = 0; t < SEQ; ++t) {
    const float*  h0o  = h0p[t & 1];
    float*        h0n  = h0p[(t + 1) & 1];
    const float*  h1o  = h1p[t & 1];
    float*        h1n  = h1p[(t + 1) & 1];
    const bf16_t* h0bo = h0bp[t & 1];
    bf16_t*       h0bn = h0bp[(t + 1) & 1];
    const bf16_t* h1bo = h1bp[t & 1];
    bf16_t*       h1bn = h1bp[(t + 1) & 1];
    gru0_rz<<<g, 32, 0, stream>>>(h0bo, h0o, bWrh, bWzh, Xr, Xz, br0, bz0,
                                  rhb, zb, t);
    gru0_h <<<g, 32, 0, stream>>>(h0o, rhb, bWhh, Xh, bh0, zb, h0n, h0bn, t);
    fc_sig <<<g, 32, 0, stream>>>(h0bn, bWfc, bfc0, x1b);
    gru1_rz<<<g, 32, 0, stream>>>(x1b, h1bo, h1o, bWr1, bWz1, br1, bz1, rhb, zb);
    gru1_h <<<g, 32, 0, stream>>>(x1b, rhb, h1o, bWh1, bh1, zb, h1n, h1bn,
                                  H1bf, t);
  }

  // ---- output projection (bulk of the FLOPs, fully parallel) ----
  out_gemm<<<dim3(ROWS / 32, VOC / 80), 32, 0, stream>>>(H1bf, bWoutT, bout,
                                                         (float*)d_out);

  // ---- final hidden state -> tail of d_out ----
  copy_hfinal<<<(BATCH * HID) / 256, 256, 0, stream>>>(
      h0p[SEQ & 1], h1p[SEQ & 1], (float*)d_out + (size_t)SEQ * BATCH * VOC);
}
